// Attention_3453153706649
// MI455X (gfx1250) — matrix operand order repaired and checked by reading, not compiled, not run on hardware
//
#include <hip/hip_runtime.h>

// ---------------------------------------------------------------------------
// MI455X (gfx1250, wave32) biased, gated multi-head attention.
// All GEMMs on v_wmma_f32_16x16x32_f16 (f16 in, f32 accumulate).
// GEMM: 64x32 register-blocked macro-tile per wave (4x2 fragments, 8 WMMA/K-step).
// ---------------------------------------------------------------------------

typedef __attribute__((ext_vector_type(16))) _Float16 v16h;
typedef __attribute__((ext_vector_type(8)))  _Float16 v8h;
typedef __attribute__((ext_vector_type(8)))  float    v8f;

union AFrag { v16h v; v8h h[2]; };

#define WMMA_F16(a, b, c) \
  __builtin_amdgcn_wmma_f32_16x16x32_f16(false, (a), false, (b), (short)0, (c), false, false)

constexpr int B_  = 2;
constexpr int L_  = 1024;
constexpr int E_  = 1024;
constexpr int H_  = 16;
constexpr int HW_ = 64;
constexpr int K_  = 1024;   // inner dim of all projections
constexpr int SLD = 1032;   // padded LDS row stride (halfs): 16B aligned, bank-conflict-free

// ------------------------- f32 -> f16 conversion ---------------------------
__global__ void cvt_f32_f16(const float* __restrict__ s, _Float16* __restrict__ d, int n) {
  int i = blockIdx.x * blockDim.x + threadIdx.x;
  if (i < n) d[i] = (_Float16)s[i];
}

// ----------------- 64x32 register-blocked WMMA GEMM ------------------------
// C[m][n] = sum_k A[m][k] * W[n][k]   (A: Mx1024, W: Nx1024, both f16 row-major)
// Each wave: 4 (M) x 2 (N) fragments = 64x32 macro-tile, 8 WMMA per K-step.
// mode 0: scatter QKV (Q scaled by hw^-0.5, V stored transposed)
// mode 1: G = sigmoid(C + b_g)  (f32)
// mode 2: out = C + b_o         (f32)
__global__ __launch_bounds__(256)
void gemm64x32(const _Float16* __restrict__ A, const _Float16* __restrict__ W,
               int N, int mode, const float* __restrict__ bvec,
               float* __restrict__ fout,
               _Float16* __restrict__ Qh, _Float16* __restrict__ Kh,
               _Float16* __restrict__ Vth)
{
  const int lane = threadIdx.x & 31;
  const int wv   = threadIdx.x >> 5;
  const int ln16 = lane & 15;
  const int hi   = lane >> 4;

  const int tilesN = N >> 5;                       // 32-wide macro-cols
  const int tileId = blockIdx.x * 8 + wv;
  const int tm = tileId / tilesN;
  const int tn = tileId - tm * tilesN;
  const int m0 = tm << 6, n0 = tn << 5;            // 64x32 macro-tile origin

  const _Float16* Arp[4];
  const _Float16* Wrp[2];
#pragma unroll
  for (int mi = 0; mi < 4; ++mi) Arp[mi] = A + (size_t)(m0 + mi * 16 + ln16) * K_;
#pragma unroll
  for (int ni = 0; ni < 2; ++ni) Wrp[ni] = W + (size_t)(n0 + ni * 16 + ln16) * K_;

  v8f acc[4][2];
#pragma unroll
  for (int mi = 0; mi < 4; ++mi)
#pragma unroll
    for (int ni = 0; ni < 2; ++ni) acc[mi][ni] = v8f{};

#pragma unroll 2
  for (int k0 = 0; k0 < K_; k0 += 32) {
    AFrag af[4], bf[2];
#pragma unroll
    for (int mi = 0; mi < 4; ++mi) {
      // A k order used here (contiguous, same as B): elems 0..15 <- K = k0+16*hi .. +15
      af[mi].h[0] = *(const v8h*)(Arp[mi] + k0 + 16 * hi);
      af[mi].h[1] = *(const v8h*)(Arp[mi] + k0 + 16 * hi + 8);
    }
#pragma unroll
    for (int ni = 0; ni < 2; ++ni) {
      // B k order used here (contiguous, same as A): elems 0..15 <- K = k0+16*hi .. +15 (col n -> row of W)
      bf[ni].h[0] = *(const v8h*)(Wrp[ni] + k0 + 16 * hi);
      bf[ni].h[1] = *(const v8h*)(Wrp[ni] + k0 + 16 * hi + 8);
    }
#pragma unroll
    for (int mi = 0; mi < 4; ++mi)
#pragma unroll
      for (int ni = 0; ni < 2; ++ni)
        acc[mi][ni] = WMMA_F16(af[mi].v, bf[ni].v, acc[mi][ni]);
  }

  // ----------------------------- epilogue ----------------------------------
#pragma unroll
  for (int mi = 0; mi < 4; ++mi) {
#pragma unroll
    for (int ni = 0; ni < 2; ++ni) {
      const int n = n0 + ni * 16 + ln16;           // output column for this lane
      if (mode == 0) {
        const int hh = n / 192, r = n % 192;       // head, slot within 3*hw
#pragma unroll
        for (int r8 = 0; r8 < 8; ++r8) {
          const int m  = m0 + mi * 16 + r8 + 8 * hi;   // row in [0, B*L)
          const int bb = m >> 10, l = m & 1023;
          const float v = acc[mi][ni][r8];
          if (r < 64)
            Qh[(((size_t)bb * H_ + hh) * L_ + l) * HW_ + r] = (_Float16)(v * 0.125f);
          else if (r < 128)
            Kh[(((size_t)bb * H_ + hh) * L_ + l) * HW_ + (r - 64)] = (_Float16)v;
          else
            Vth[(((size_t)bb * H_ + hh) * HW_ + (r - 128)) * L_ + l] = (_Float16)v;
        }
      } else if (mode == 1) {
        const float bg = bvec[n];
#pragma unroll
        for (int r8 = 0; r8 < 8; ++r8) {
          const int m = m0 + mi * 16 + r8 + 8 * hi;
          const float v = acc[mi][ni][r8] + bg;
          fout[(size_t)m * E_ + n] = 1.0f / (1.0f + __expf(-v));
        }
      } else {
        const float bo = bvec[n];
#pragma unroll
        for (int r8 = 0; r8 < 8; ++r8) {
          const int m = m0 + mi * 16 + r8 + 8 * hi;
          fout[(size_t)m * E_ + n] = acc[mi][ni][r8] + bo;
        }
      }
    }
  }
}

// ------------------------------ attention ----------------------------------
// One block per (b, h, 16-row q tile). 8 waves / 256 threads.
__global__ __launch_bounds__(256)
void attn16(const _Float16* __restrict__ Qh, const _Float16* __restrict__ Kh,
            const _Float16* __restrict__ Vth, const float* __restrict__ bias,
            float* __restrict__ attnO, float* __restrict__ Ybuf)
{
  __shared__ __align__(16) _Float16 Sh[16][SLD];   // logits -> exp -> normalized P
  __shared__ float red[16][16];
  __shared__ float rowmax[16];
  __shared__ float rowinv[16];
  __shared__ float Yred[16][64];

  const int tid  = threadIdx.x;
  const int lane = tid & 31;
  const int wv   = tid >> 5;
  const int ln16 = lane & 15;
  const int hi   = lane >> 4;

  const int blk = blockIdx.x;             // B * H * (L/16)
  const int b   = blk / (H_ * 64);
  const int rm  = blk - b * H_ * 64;
  const int h   = rm >> 6;
  const int q0  = (rm & 63) << 4;

  const _Float16* Qbase = Qh  + (((size_t)b * H_ + h) * L_) * HW_;
  const _Float16* Kbase = Kh  + (((size_t)b * H_ + h) * L_) * HW_;
  const _Float16* Vbase = Vth + (((size_t)b * H_ + h) * HW_) * L_;

  // Q fragments (identical q rows for every wave; L2 serves the rebroadcast)
  AFrag aq0, aq1;
  {
    const _Float16* Qr = Qbase + (size_t)(q0 + ln16) * HW_;
    aq0.h[0] = *(const v8h*)(Qr + 16 * hi);
    aq0.h[1] = *(const v8h*)(Qr + 16 * hi + 8);
    aq1.h[0] = *(const v8h*)(Qr + 32 + 16 * hi);
    aq1.h[1] = *(const v8h*)(Qr + 32 + 16 * hi + 8);
  }

  // ---- S = Q K^T + bias -> LDS (f16 logits) ----
  for (int kt = wv; kt < 64; kt += 8) {
    const int k0 = kt << 4;
    const _Float16* Kr = Kbase + (size_t)(k0 + ln16) * HW_;
    AFrag b0, b1;
    b0.h[0] = *(const v8h*)(Kr + 16 * hi);            // channels 0..31
    b0.h[1] = *(const v8h*)(Kr + 16 * hi + 8);
    b1.h[0] = *(const v8h*)(Kr + 32 + 16 * hi);       // channels 32..63
    b1.h[1] = *(const v8h*)(Kr + 32 + 16 * hi + 8);
    v8f c = {};
    c = WMMA_F16(aq0.v, b0.v, c);
    c = WMMA_F16(aq1.v, b1.v, c);
    // bias[b][q0+M][k0+n][h]
    const float* bp = bias + (((size_t)b * L_ + q0) * L_ + (k0 + ln16)) * H_ + h;
#pragma unroll
    for (int r = 0; r < 8; ++r) {
      const int M = r + 8 * hi;
      const float v = c[r] + bp[(size_t)M * (L_ * H_)];
      Sh[M][k0 + ln16] = (_Float16)v;
    }
  }
  __syncthreads();

  // ---- softmax over k (f32 math on f16 logits) ----
  const int row = tid & 15;
  const int kk  = tid >> 4;
  {
    float m = -1e30f;
    for (int j = kk * 64; j < kk * 64 + 64; ++j)
      m = fmaxf(m, (float)Sh[row][j]);
    red[kk][row] = m;
  }
  __syncthreads();
  if (tid < 16) {
    float mm = red[0][tid];
    for (int i = 1; i < 16; ++i) mm = fmaxf(mm, red[i][tid]);
    rowmax[tid] = mm;
  }
  __syncthreads();
  {
    const float rmax = rowmax[row];
    float s = 0.f;
    for (int j = kk * 64; j < kk * 64 + 64; ++j) {
      const float p = __expf((float)Sh[row][j] - rmax);
      Sh[row][j] = (_Float16)p;
      s += p;
    }
    red[kk][row] = s;
  }
  __syncthreads();
  if (tid < 16) {
    float ss = red[0][tid];
    for (int i = 1; i < 16; ++i) ss += red[i][tid];
    rowinv[tid] = 1.0f / ss;
  }
  __syncthreads();

  // ---- normalize, emit attn (B, Lk, H, Lq): q-contiguous 64B group stores ----
  {
    float* aO = attnO + (((size_t)b * L_) * H_ + h) * L_ + q0;  // + k*(H_*L_) + q
    const int qw = tid & 15;
    const int kb = tid >> 4;
    const float inv = rowinv[qw];
    for (int i = 0; i < 64; ++i) {
      const int k = kb + (i << 4);
      const float p = (float)Sh[qw][k] * inv;
      Sh[qw][k] = (_Float16)p;                 // keep normalized P for PV
      aO[(size_t)k * (H_ * L_) + qw] = p;
    }
  }
  __syncthreads();

  // ---- Y = P V : waves split (n-tile of 16 channels) x (K half of 512) ----
  const int c0t    = (wv & 3) << 4;
  const int kstart = (wv >> 2) << 9;
  v8f acc = {};
  for (int k0 = kstart; k0 < kstart + 512; k0 += 32) {
    AFrag a, vb;
    a.h[0] = *(const v8h*)&Sh[ln16][k0 + 16 * hi];
    a.h[1] = *(const v8h*)&Sh[ln16][k0 + 16 * hi + 8];
    const _Float16* Vr = Vbase + (size_t)(c0t + ln16) * L_ + k0 + 16 * hi;  // V transposed
    vb.h[0] = *(const v8h*)(Vr);
    vb.h[1] = *(const v8h*)(Vr + 8);
    acc = WMMA_F16(a.v, vb.v, acc);
  }
  if (wv >= 4) {
#pragma unroll
    for (int r = 0; r < 8; ++r) Yred[r + 8 * hi][c0t + ln16] = acc[r];
  }
  __syncthreads();
  if (wv < 4) {
    float* Yr = Ybuf + ((size_t)b * L_ + q0) * E_ + h * HW_ + c0t + ln16;
#pragma unroll
    for (int r = 0; r < 8; ++r) {
      const int M = r + 8 * hi;
      Yr[(size_t)M * E_] = acc[r] + Yred[M][c0t + ln16];
    }
  }
}

// ------------------------------ gating -------------------------------------
__global__ void gatemul(const float* __restrict__ G, const float* __restrict__ Y,
                        _Float16* __restrict__ O, int n) {
  int i = blockIdx.x * blockDim.x + threadIdx.x;
  if (i < n) O[i] = (_Float16)(G[i] * Y[i]);
}

// ------------------------------ launcher -----------------------------------
extern "C" void kernel_launch(void* const* d_in, const int* in_sizes, int n_in,
                              void* d_out, int out_size, void* d_ws, size_t ws_size,
                              hipStream_t stream)
{
  const float* x      = (const float*)d_in[0];
  const float* bias   = (const float*)d_in[1];
  const float* w_proj = (const float*)d_in[2];
  const float* w_o    = (const float*)d_in[3];
  const float* b_o    = (const float*)d_in[4];
  const float* w_g    = (const float*)d_in[5];
  const float* b_g    = (const float*)d_in[6];

  char*  ws  = (char*)d_ws;
  size_t off = 0;
  auto alloc = [&](size_t bytes) -> void* {
    void* p = ws + off;
    off += (bytes + 255) & ~size_t(255);
    return p;
  };
  _Float16* Xh  = (_Float16*)alloc((size_t)B_ * L_ * E_ * 2);       // 4MB
  _Float16* Wph = (_Float16*)alloc((size_t)3 * E_ * E_ * 2);        // 6MB
  _Float16* Wgh = (_Float16*)alloc((size_t)E_ * E_ * 2);            // 2MB
  _Float16* Woh = (_Float16*)alloc((size_t)E_ * E_ * 2);            // 2MB
  _Float16* Qh  = (_Float16*)alloc((size_t)B_ * H_ * L_ * HW_ * 2); // 4MB
  _Float16* Kh  = (_Float16*)alloc((size_t)B_ * H_ * L_ * HW_ * 2); // 4MB
  _Float16* Vth = (_Float16*)alloc((size_t)B_ * H_ * HW_ * L_ * 2); // 4MB (transposed)
  float*    G   = (float*)alloc((size_t)B_ * L_ * E_ * 4);          // 8MB
  float*    Yb  = (float*)alloc((size_t)B_ * L_ * E_ * 4);          // 8MB
  _Float16* Ygh = (_Float16*)alloc((size_t)B_ * L_ * E_ * 2);       // 4MB

  float* yOut  = (float*)d_out;                                  // (B,L,E)
  float* attnO = (float*)d_out + (size_t)B_ * L_ * E_;           // (B,Lk,H,Lq)

  const int nX = B_ * L_ * E_, nWp = 3 * E_ * E_, nW = E_ * E_;
  cvt_f32_f16<<<(nX  + 255) / 256, 256, 0, stream>>>(x,      Xh,  nX);
  cvt_f32_f16<<<(nWp + 255) / 256, 256, 0, stream>>>(w_proj, Wph, nWp);
  cvt_f32_f16<<<(nW  + 255) / 256, 256, 0, stream>>>(w_g,    Wgh, nW);
  cvt_f32_f16<<<(nW  + 255) / 256, 256, 0, stream>>>(w_o,    Woh, nW);

  // QKV projection: M=2048 (32 macro-rows), N=3072 (96 macro-cols) -> 3072 wave-tiles / 8
  gemm64x32<<<(32 * 96) / 8, 256, 0, stream>>>(Xh, Wph, 3 * E_, 0, nullptr, nullptr, Qh, Kh, Vth);
  // gate projection: M=2048, N=1024 (32 macro-cols) -> 1024 wave-tiles / 8
  gemm64x32<<<(32 * 32) / 8, 256, 0, stream>>>(Xh, Wgh, E_, 1, b_g, G, nullptr, nullptr, nullptr);
  // attention: one block per (b,h,q-tile)
  attn16<<<B_ * H_ * (L_ / 16), 256, 0, stream>>>(Qh, Kh, Vth, bias, attnO, Yb);
  // gating
  gatemul<<<(nX + 255) / 256, 256, 0, stream>>>(G, Yb, Ygh, nX);
  // output projection: M=2048, N=1024
  gemm64x32<<<(32 * 32) / 8, 256, 0, stream>>>(Ygh, Woh, E_, 2, b_o, yOut, nullptr, nullptr, nullptr);
}
